// CoLAMoELayer_18279380812215
// MI455X (gfx1250) — matrix operand order repaired and checked by reading, not compiled, not run on hardware
//
#include <hip/hip_runtime.h>
#include <hip/hip_bf16.h>

// ---------------- problem constants (from reference) ----------------
constexpr int BATCH = 4;
constexpr int SEQ   = 2048;
constexpr int TOKS  = BATCH * SEQ;   // 8192 tokens
constexpr int DIM   = 2048;          // model dim (in == out)
constexpr int NEXP  = 8;             // experts
// TOPK = 2 (hardcoded in gating kernel)

// ---------------- GEMM tiling ----------------
constexpr int BM = 256;    // tokens per block tile
constexpr int BN = 128;    // output cols per block tile
constexpr int BK = 32;     // K step == wmma K
constexpr int NTHR = 512;  // 16 waves

typedef __attribute__((ext_vector_type(16))) __bf16 v16bf;
typedef __attribute__((ext_vector_type(8)))  __bf16 v8bf;
typedef __attribute__((ext_vector_type(8)))  float  v8f;

// =====================================================================
// Kernel 1: gating.  One wave32 per token.
//   logits[t,e] = sum_d x[t,d]*gate_w[e,d];  top-2 + softmax;
//   scatter (token, weight) into per-expert buckets via atomics.
//   Each output element later receives exactly 2 fp32 atomic adds,
//   which is commutative -> bitwise deterministic result.
// =====================================================================
__global__ __launch_bounds__(256)
void moe_gate_kernel(const float* __restrict__ x,
                     const float* __restrict__ gate_w,
                     int*   __restrict__ counts,
                     int*   __restrict__ btok,
                     float* __restrict__ bw)
{
    const int wave = threadIdx.x >> 5;
    const int lane = threadIdx.x & 31;
    const int t    = blockIdx.x * 8 + wave;
    if (t >= TOKS) return;

    float acc[NEXP];
#pragma unroll
    for (int e = 0; e < NEXP; ++e) acc[e] = 0.0f;

    const float* xr = x + (size_t)t * DIM;
    for (int d = lane * 4; d < DIM; d += 32 * 4) {
        const float4 xv = *(const float4*)(xr + d);
#pragma unroll
        for (int e = 0; e < NEXP; ++e) {
            const float4 gv = *(const float4*)(gate_w + (size_t)e * DIM + d);
            acc[e] += xv.x * gv.x + xv.y * gv.y + xv.z * gv.z + xv.w * gv.w;
        }
    }
#pragma unroll
    for (int off = 16; off >= 1; off >>= 1) {
#pragma unroll
        for (int e = 0; e < NEXP; ++e)
            acc[e] += __shfl_xor(acc[e], off, 32);
    }

    if (lane == 0) {
        // top-2 (ties -> lower index, matching jax.lax.top_k)
        int i0 = 0; float v0 = acc[0];
#pragma unroll
        for (int e = 1; e < NEXP; ++e)
            if (acc[e] > v0) { v0 = acc[e]; i0 = e; }
        int i1 = -1; float v1 = -3.4e38f;
#pragma unroll
        for (int e = 0; e < NEXP; ++e)
            if (e != i0 && acc[e] > v1) { v1 = acc[e]; i1 = e; }
        const float e1 = __expf(v1 - v0);
        const float inv = 1.0f / (1.0f + e1);

        int p0 = atomicAdd(&counts[i0], 1);
        btok[i0 * TOKS + p0] = t;
        bw  [i0 * TOKS + p0] = inv;
        int p1 = atomicAdd(&counts[i1], 1);
        btok[i1 * TOKS + p1] = t;
        bw  [i1 * TOKS + p1] = e1 * inv;
    }
}

// =====================================================================
// Kernel 2: grouped expert GEMM, bf16 WMMA + fp32 accumulate.
//   out[tok] += w * (x[tok] @ W_e + b_e) for the tokens routed to e.
//   256x128x32 block tile, 512 threads (16 waves), each wave: 64x32
//   output = 4x2 v_wmma_f32_16x16x32_bf16 per K-step.
//   Software-pipelined: next K-step's global loads are issued before
//   the WMMA phase; conversions/LDS stores consume them next iteration.
// =====================================================================
__global__ __launch_bounds__(NTHR)
void moe_expert_gemm_kernel(const float* __restrict__ x,
                            const float* __restrict__ expert_w,
                            const float* __restrict__ expert_b,
                            const int*   __restrict__ counts,
                            const int*   __restrict__ btok,
                            const float* __restrict__ bw,
                            float* __restrict__ out)
{
    const int e   = blockIdx.z;
    const int n_e = counts[e];
    const int m0  = blockIdx.y * BM;
    if (m0 >= n_e) return;               // uniform per block
    const int n0  = blockIdx.x * BN;

    __shared__ __align__(64) __bf16 Asl[BM][BK];   // token rows, K-contiguous (16 KB)
    __shared__ __align__(64) __bf16 Bsl[BN][BK];   // W tile transposed [n][k]  (8 KB)
    __shared__ int   tokS[BM];
    __shared__ float wS[BM];

    const int tid = threadIdx.x;

    if (tid < BM) {
        const int r = m0 + tid;
        const bool ok = (r < n_e);
        tokS[tid] = ok ? btok[e * TOKS + r] : -1;
        wS[tid]   = ok ? bw  [e * TOKS + r] : 0.0f;
    }
    __syncthreads();

    const float* Wbase = expert_w + (size_t)e * DIM * DIM;

    // ---- staging maps ----
    // A: 8 threads per row (float4 along K); rows arow + i*64, i=0..3
    const int arow = tid >> 3;           // 0..63
    const int acg  = (tid & 7) * 4;      // K col group 0..28
    int atok[4];
#pragma unroll
    for (int i = 0; i < 4; ++i) atok[i] = tokS[arow + i * 64];

    // B: one (k-pair, 4-col) slot per thread: rows 2*bkp, 2*bkp+1; cols bcg..bcg+3
    const int bkp = tid >> 5;            // 0..15
    const int bcg = (tid & 31) * 4;      // 0..124

    float4 aPre[4];
    float4 bPre[2];

    auto loadA = [&](int k0) {
#pragma unroll
        for (int i = 0; i < 4; ++i) {
            float4 v = make_float4(0.f, 0.f, 0.f, 0.f);
            if (atok[i] >= 0)
                v = *(const float4*)(x + (size_t)atok[i] * DIM + k0 + acg);
            aPre[i] = v;
        }
    };
    auto loadB = [&](int k0) {
        const float* p0 = Wbase + (size_t)(k0 + 2 * bkp) * DIM + n0 + bcg;
        bPre[0] = *(const float4*)p0;
        bPre[1] = *(const float4*)(p0 + DIM);
    };
    auto storeA = [&]() {
#pragma unroll
        for (int i = 0; i < 4; ++i) {
            union { __bf16 h[4]; unsigned long long u; } q;
            q.h[0] = (__bf16)aPre[i].x; q.h[1] = (__bf16)aPre[i].y;
            q.h[2] = (__bf16)aPre[i].z; q.h[3] = (__bf16)aPre[i].w;
            *(unsigned long long*)&Asl[arow + i * 64][acg] = q.u;   // ds_store_b64
        }
    };
    auto storeB = [&]() {
        const float f0[4] = { bPre[0].x, bPre[0].y, bPre[0].z, bPre[0].w };
        const float f1[4] = { bPre[1].x, bPre[1].y, bPre[1].z, bPre[1].w };
#pragma unroll
        for (int j = 0; j < 4; ++j) {
            union { __bf16 h[2]; unsigned int u; } p;   // pack (k, k+1) pair
            p.h[0] = (__bf16)f0[j]; p.h[1] = (__bf16)f1[j];
            *(unsigned int*)&Bsl[bcg + j][2 * bkp] = p.u;           // ds_store_b32
        }
    };

    // ---- wave tiling: 16 waves in 4(m) x 4(n); wave tile 64x32 ----
    const int wave = tid >> 5;     // 0..15
    const int lane = tid & 31;
    const int wm   = wave >> 2;    // 0..3 : rows wm*64
    const int wn   = wave & 3;     // 0..3 : cols wn*32
    const int half = lane >> 4;    // 0/1
    const int l16  = lane & 15;

    v8f acc[4][2];
#pragma unroll
    for (int mi = 0; mi < 4; ++mi)
#pragma unroll
        for (int ni = 0; ni < 2; ++ni) {
            v8f z = {0.f,0.f,0.f,0.f,0.f,0.f,0.f,0.f};
            acc[mi][ni] = z;
        }

    loadA(0);
    loadB(0);

    for (int k0 = 0; k0 < DIM; k0 += BK) {
        storeA();
        storeB();
        __syncthreads();

        // issue next K-step's global loads before the WMMA phase
        if (k0 + BK < DIM) { loadA(k0 + BK); loadB(k0 + BK); }

        // B fragments (reused across the 4 m-subtiles)
        v16bf bfr[2];
#pragma unroll
        for (int ni = 0; ni < 2; ++ni) {
            const int cb = wn * 32 + ni * 16 + l16;
            bfr[ni] = *(const v16bf*)&Bsl[cb][half * 16];
        }
#pragma unroll
        for (int mi = 0; mi < 4; ++mi) {
            // A fragment: lanes<16 carry K 0-15, lanes>=16 carry K 16-31
            const int rbase = wm * 64 + mi * 16 + l16;
            union { v16bf v; v8bf h[2]; } afr;
            const __bf16* ap = &Asl[rbase][half * 16];
            afr.h[0] = *(const v8bf*)(ap);
            afr.h[1] = *(const v8bf*)(ap + 8);
#pragma unroll
            for (int ni = 0; ni < 2; ++ni)
                acc[mi][ni] = __builtin_amdgcn_wmma_f32_16x16x32_bf16(
                    false, afr.v, false, bfr[ni], (short)0, acc[mi][ni], false, false);
        }
        __syncthreads();
    }

    // ---- epilogue: out[tok, col] += w * (acc + bias) ----
#pragma unroll
    for (int ni = 0; ni < 2; ++ni) {
        const int col  = n0 + wn * 32 + ni * 16 + l16;
        const float bias = expert_b[(size_t)e * DIM + col];
#pragma unroll
        for (int mi = 0; mi < 4; ++mi) {
            const int mbase = wm * 64 + mi * 16 + half * 8;
#pragma unroll
            for (int r = 0; r < 8; ++r) {
                const int mrow = mbase + r;
                const int tok  = tokS[mrow];
                if (tok >= 0) {
                    const float v = wS[mrow] * (acc[mi][ni][r] + bias);
                    atomicAdd(out + (size_t)tok * DIM + col, v);
                }
            }
        }
    }
}

// =====================================================================
// Host-side launch
// =====================================================================
extern "C" void kernel_launch(void* const* d_in, const int* in_sizes, int n_in,
                              void* d_out, int out_size, void* d_ws, size_t ws_size,
                              hipStream_t stream)
{
    const float* x        = (const float*)d_in[0];   // [T, D]
    const float* gate_w   = (const float*)d_in[1];   // [E, D]
    const float* expert_w = (const float*)d_in[2];   // [E, D, D]
    const float* expert_b = (const float*)d_in[3];   // [E, D]
    float* out = (float*)d_out;                      // [T, D]

    // workspace layout: counts (64B) | btok (E*T ints) | bw (E*T floats)
    int*   counts = (int*)d_ws;
    int*   btok   = (int*)((char*)d_ws + 64);
    float* bw     = (float*)((char*)d_ws + 64 + (size_t)NEXP * TOKS * sizeof(int));

    hipMemsetAsync(d_ws, 0, 64, stream);                                    // counts = 0
    hipMemsetAsync(d_out, 0, (size_t)TOKS * DIM * sizeof(float), stream);   // out = 0

    moe_gate_kernel<<<TOKS / 8, 256, 0, stream>>>(x, gate_w, counts, btok, bw);

    dim3 grid(DIM / BN, TOKS / BM, NEXP);   // (16, 32, 8); empty tiles early-exit
    moe_expert_gemm_kernel<<<grid, NTHR, 0, stream>>>(
        x, expert_w, expert_b, counts, btok, bw, out);
}
